// CrossTransformerBlock_39213051412830
// MI455X (gfx1250) — matrix operand order repaired and checked by reading, not compiled, not run on hardware
//
#include <hip/hip_runtime.h>
#include <hip/hip_bf16.h>

typedef __attribute__((ext_vector_type(16))) _Float16 v16h;
typedef __attribute__((ext_vector_type(8)))  float    v8f;
typedef __attribute__((ext_vector_type(4)))  float    f4;
typedef __attribute__((ext_vector_type(4)))  unsigned int u4;

#define NB   4
#define NQ   4096
#define NPTS 2048
#define DIN  256
#define DIM  128
#define KNN  16

// ---------------- workspace byte offsets ----------------
static const size_t WS_KFEAT = 0;                               // 4*2048*128 f32 = 4 MB
static const size_t WS_VFEAT = 4194304;                         // 4 MB
static const size_t WS_IDX   = 8388608;                         // 4*4096*16 i32 = 1 MB
static const size_t WS_QVEC  = 9437184;                         // 4*128 f32
static const size_t WS_VG    = WS_QVEC + 2048;
static const size_t WS_GLOG  = WS_VG   + 2048;
static const size_t WS_PACK_KS  = WS_GLOG + 2048;               // 64 KB (256x128 f16 frags)
static const size_t WS_PACK_VS  = WS_PACK_KS + 65536;           // 64 KB
static const size_t WS_PACK_DIM = WS_PACK_VS + 65536;           // dw2,gw1,gw2: 3 x 32 KB

// ---------------- LDS byte offsets (main kernel, dynamic) ----------------
#define LDS_W    0         // 3 x 32768 packed dim weights (dw2, gw1, gw2)
#define LDS_POS  98304     // 8 waves x 16x128 f32 scratch (64 KB)
#define LDS_DW1  163840
#define LDS_DB1  165376
#define LDS_DB2  165888
#define LDS_GB1  166400
#define LDS_GB2  166912
#define LDS_Q    167424
#define LDS_GL   167936
#define LDS_VG   168448
#define LDS_IDX  168960
#define LDS_TOTAL 169472

// =====================================================================
// Kernel P: repack weights into wave32 WMMA B-fragment order (f16).
// B frag: lane L holds col n = 16*nt + (L&15); element e -> k = 32*kf + 8*(L>>4) + e + (e>=8 ? 8 : 0)
// =====================================================================
__global__ void pack_weights_kernel(const float* __restrict__ w_ks,
                                    const float* __restrict__ w_vs,
                                    const float* __restrict__ dw2,
                                    const float* __restrict__ gw1,
                                    const float* __restrict__ gw2,
                                    char* __restrict__ ws) {
  int mat = blockIdx.x;
  const float* src = (mat == 0) ? w_ks : (mat == 1) ? w_vs
                   : (mat == 2) ? dw2  : (mat == 3) ? gw1 : gw2;
  size_t dstoff = (mat == 0) ? WS_PACK_KS
                : (mat == 1) ? WS_PACK_VS
                : (WS_PACK_DIM + (size_t)(mat - 2) * 32768u);
  int kfc = (mat < 2) ? 8 : 4;              // K = 256 or 128
  int total = 8 * kfc * 32;                 // (nt, kf, lane)
  _Float16* dst = (_Float16*)(ws + dstoff);
  for (int t = threadIdx.x; t < total; t += blockDim.x) {
    int lane = t & 31;
    int rest = t >> 5;
    int kf = rest % kfc;
    int nt = rest / kfc;
    int grp = lane >> 4;
    int n = 16 * nt + (lane & 15);
#pragma unroll
    for (int e = 0; e < 16; e++) {
      int k = 32 * kf + 8 * grp + e + ((e >= 8) ? 8 : 0);
      dst[(size_t)t * 16 + e] = (_Float16)src[k * 128 + n];
    }
  }
}

// =====================================================================
// Kernel 0: k_feat = points @ w_ks, v_feat = points @ w_vs (f16 WMMA, f32 out)
// one wave per 16-row tile, both outputs share the A fragments
// =====================================================================
__global__ void feat_gemm_kernel(const float* __restrict__ points,
                                 char* __restrict__ ws) {
  const int lane = threadIdx.x & 31;
  const int wave = threadIdx.x >> 5;
  const int lanelo = lane & 15, grp = lane >> 4;
  const int tile = blockIdx.x * 8 + wave;          // 0..511
  const size_t row = (size_t)tile * 16 + lanelo;   // folded (b*N) row
  const float* pr = points + row * DIN;

  v16h aF[8];
#pragma unroll
  for (int kf = 0; kf < 8; kf++) {
    int c0 = 32 * kf + grp * 8;
    f4 a0 = *(const f4*)(pr + c0);
    f4 a1 = *(const f4*)(pr + c0 + 4);
    f4 a2 = *(const f4*)(pr + c0 + 16);
    f4 a3 = *(const f4*)(pr + c0 + 20);
#pragma unroll
    for (int e = 0; e < 4; e++) {
      aF[kf][e]      = (_Float16)a0[e];
      aF[kf][e + 4]  = (_Float16)a1[e];
      aF[kf][e + 8]  = (_Float16)a2[e];
      aF[kf][e + 12] = (_Float16)a3[e];
    }
  }

#pragma unroll
  for (int mt = 0; mt < 2; mt++) {
    const _Float16* pk = (const _Float16*)(ws + (mt == 0 ? WS_PACK_KS : WS_PACK_VS));
    float* dst = (float*)(ws + (mt == 0 ? WS_KFEAT : WS_VFEAT));
#pragma unroll
    for (int nt = 0; nt < 8; nt++) {
      v8f acc;
#pragma unroll
      for (int i = 0; i < 8; i++) acc[i] = 0.f;
#pragma unroll
      for (int kf = 0; kf < 8; kf++) {
        v16h b = *(const v16h*)(pk + ((size_t)((nt * 8 + kf) * 32 + lane)) * 16);
        acc = __builtin_amdgcn_wmma_f32_16x16x32_f16(false, aF[kf], false, b,
                                                     (short)0, acc, false, false);
      }
#pragma unroll
      for (int r = 0; r < 8; r++) {
        size_t orow = (size_t)tile * 16 + r + grp * 8;
        dst[orow * 128 + 16 * nt + lanelo] = acc[r];
      }
    }
  }
}

// =====================================================================
// Kernel 1: KNN (top-16 smallest squared distances), xyz cached in LDS
// =====================================================================
__global__ void knn_kernel(const float* __restrict__ xyz_q,
                           const float* __restrict__ xyz,
                           char* __restrict__ ws) {
  __shared__ float sx[NPTS * 3];
  const int bb = blockIdx.y;
  const float* xb = xyz + (size_t)bb * NPTS * 3;
  for (int i = threadIdx.x; i < NPTS * 3; i += blockDim.x) sx[i] = xb[i];
  __syncthreads();

  const int q = blockIdx.x * blockDim.x + threadIdx.x;
  const float* qp = xyz_q + ((size_t)bb * NQ + q) * 3;
  float qx = qp[0], qy = qp[1], qz = qp[2];

  float bd[KNN];
  int bi[KNN];
#pragma unroll
  for (int j = 0; j < KNN; j++) { bd[j] = 3.4e38f; bi[j] = 0; }

  for (int n = 0; n < NPTS; n++) {
    float dx = qx - sx[n * 3], dy = qy - sx[n * 3 + 1], dz = qz - sx[n * 3 + 2];
    float d = dx * dx + dy * dy + dz * dz;
    if (d < bd[KNN - 1]) {
      bd[KNN - 1] = d; bi[KNN - 1] = n;
#pragma unroll
      for (int j = KNN - 1; j > 0; --j) {
        if (bd[j] < bd[j - 1]) {
          float td = bd[j]; bd[j] = bd[j - 1]; bd[j - 1] = td;
          int ti = bi[j];  bi[j] = bi[j - 1];  bi[j - 1] = ti;
        }
      }
    }
  }
  int* idst = (int*)(ws + WS_IDX) + ((size_t)bb * NQ + q) * KNN;
#pragma unroll
  for (int j = 0; j < KNN; j++) idst[j] = bi[j];
}

// =====================================================================
// Kernel 2: per-batch globals: qvec = lat@w_qs, vg = lat@w_vg,
//           glog = MLP(qvec - lat@w_kg)  (global-slot attention logits)
// =====================================================================
__global__ void globals_kernel(const float* __restrict__ lat,
                               const float* __restrict__ w_qs,
                               const float* __restrict__ w_kg,
                               const float* __restrict__ w_vg,
                               const float* __restrict__ gw1,
                               const float* __restrict__ gb1,
                               const float* __restrict__ gw2,
                               const float* __restrict__ gb2,
                               char* __restrict__ ws) {
  __shared__ float sq[128], skg[128], sh[128];
  const int bb = blockIdx.x, c = threadIdx.x;
  const float* lb = lat + (size_t)bb * DIN;
  float aq = 0.f, ak = 0.f, av = 0.f;
  for (int k = 0; k < DIN; k++) {
    float l = lb[k];
    aq += l * w_qs[k * 128 + c];
    ak += l * w_kg[k * 128 + c];
    av += l * w_vg[k * 128 + c];
  }
  sq[c] = aq; skg[c] = ak;
  ((float*)(ws + WS_QVEC))[bb * 128 + c] = aq;
  ((float*)(ws + WS_VG))[bb * 128 + c] = av;
  __syncthreads();
  float h = gb1[c];
  for (int k = 0; k < 128; k++) h += (sq[k] - skg[k]) * gw1[k * 128 + c];
  sh[c] = fmaxf(h, 0.f);
  __syncthreads();
  float g = gb2[c];
  for (int k = 0; k < 128; k++) g += sh[k] * gw2[k * 128 + c];
  ((float*)(ws + WS_GLOG))[bb * 128 + c] = g;
}

// =====================================================================
// Kernel 3: fused main — one wave per query.
//   pos = relu(Δ@dw1 + db1)@dw2 + db2        (layer1 VALU K=3, layer2 WMMA)
//   h   = relu((q - k + pos)@gw1 + gb1)      (WMMA)
//   lgt = h@gw2 + gb2                        (WMMA)
//   softmax over 16 neighbor slots + global slot (per channel),
//   res = sum attn*(v+pos) + attn_g*vg
// =====================================================================
__global__ void cross_attn_kernel(const float* __restrict__ xyz_q,
                                  const float* __restrict__ xyz,
                                  const float* __restrict__ dw1,
                                  const float* __restrict__ db1,
                                  const float* __restrict__ db2,
                                  const float* __restrict__ gb1,
                                  const float* __restrict__ gb2,
                                  char* __restrict__ ws,
                                  float* __restrict__ out) {
  extern __shared__ char smem[];
  const int tid = threadIdx.x;
  const int lane = tid & 31, wave = tid >> 5;
  const int lanelo = lane & 15, grp = lane >> 4;
  const int bb = blockIdx.y;
  const int q = blockIdx.x * 8 + wave;

  float* dw1L = (float*)(smem + LDS_DW1);
  float* db1L = (float*)(smem + LDS_DB1);
  float* db2L = (float*)(smem + LDS_DB2);
  float* gb1L = (float*)(smem + LDS_GB1);
  float* gb2L = (float*)(smem + LDS_GB2);
  float* qvL  = (float*)(smem + LDS_Q);
  float* glL  = (float*)(smem + LDS_GL);
  float* vgL  = (float*)(smem + LDS_VG);
  int*   idxL = (int*)  (smem + LDS_IDX);
  float* posS = (float*)(smem + LDS_POS) + wave * 2048;

  // ---- block init: stage packed weights + vectors into LDS ----
  {
    u4* dstW = (u4*)smem;
    const u4* srcW = (const u4*)(ws + WS_PACK_DIM);
    for (int i = tid; i < 6144; i += 256) dstW[i] = srcW[i];   // 96 KB
    for (int i = tid; i < 384; i += 256) dw1L[i] = dw1[i];
    if (tid < 128) {
      db1L[tid] = db1[tid]; db2L[tid] = db2[tid];
      gb1L[tid] = gb1[tid]; gb2L[tid] = gb2[tid];
      qvL[tid] = ((const float*)(ws + WS_QVEC))[bb * 128 + tid];
      glL[tid] = ((const float*)(ws + WS_GLOG))[bb * 128 + tid];
      vgL[tid] = ((const float*)(ws + WS_VG))[bb * 128 + tid];
    }
  }
  __syncthreads();

  // ---- neighbor indices & position deltas ----
  const int* idxg = (const int*)(ws + WS_IDX) + ((size_t)bb * NQ + q) * KNN;
  int myidx = idxg[lanelo];                 // row m = lane&15 (dup across halves)
  if (lane < 16) idxL[wave * 16 + lane] = myidx;

  const float* qp = xyz_q + ((size_t)bb * NQ + q) * 3;
  const float* np = xyz + ((size_t)bb * NPTS + myidx) * 3;
  float d0 = qp[0] - np[0], d1 = qp[1] - np[1], d2 = qp[2] - np[2];

  // ---- pos layer 1 (K=3, VALU) packed straight into A fragments ----
  v16h aF[4];
#pragma unroll
  for (int kf = 0; kf < 4; kf++) {
#pragma unroll
    for (int e = 0; e < 16; e++) {
      int c = 32 * kf + grp * 8 + e + ((e >= 8) ? 8 : 0);
      float v = d0 * dw1L[c] + d1 * dw1L[128 + c] + d2 * dw1L[256 + c] + db1L[c];
      aF[kf][e] = (_Float16)fmaxf(v, 0.f);
    }
  }

  // ---- pos layer 2: WMMA (mat 0 = dw2), bias db2; keep in regs + LDS ----
  v8f posr[8];
#pragma unroll
  for (int nt = 0; nt < 8; nt++) {
    float bias = db2L[16 * nt + lanelo];
    v8f acc;
#pragma unroll
    for (int i = 0; i < 8; i++) acc[i] = bias;
#pragma unroll
    for (int kf = 0; kf < 4; kf++) {
      v16h b = *(const v16h*)(smem + 0 * 32768 + ((nt * 4 + kf) * 32 + lane) * 32);
      acc = __builtin_amdgcn_wmma_f32_16x16x32_f16(false, aF[kf], false, b,
                                                   (short)0, acc, false, false);
    }
    posr[nt] = acc;
#pragma unroll
    for (int r = 0; r < 8; r++)
      posS[(r + grp * 8) * 128 + 16 * nt + lanelo] = acc[r];
  }

  // ---- X = q - k + pos -> A fragments (LDS transpose for pos) ----
  const float* krow = (const float*)(ws + WS_KFEAT) + ((size_t)bb * NPTS + myidx) * 128;
  const float* prow = posS + lanelo * 128;
#pragma unroll
  for (int kf = 0; kf < 4; kf++) {
    int c0 = 32 * kf + grp * 8;
    f4 k0 = *(const f4*)(krow + c0),      k1 = *(const f4*)(krow + c0 + 4);
    f4 k2 = *(const f4*)(krow + c0 + 16), k3 = *(const f4*)(krow + c0 + 20);
    f4 p0 = *(const f4*)(prow + c0),      p1 = *(const f4*)(prow + c0 + 4);
    f4 p2 = *(const f4*)(prow + c0 + 16), p3 = *(const f4*)(prow + c0 + 20);
    f4 q0 = *(const f4*)(qvL + c0),       q1 = *(const f4*)(qvL + c0 + 4);
    f4 q2 = *(const f4*)(qvL + c0 + 16),  q3 = *(const f4*)(qvL + c0 + 20);
    f4 x0 = q0 - k0 + p0, x1 = q1 - k1 + p1, x2 = q2 - k2 + p2, x3 = q3 - k3 + p3;
#pragma unroll
    for (int e = 0; e < 4; e++) {
      aF[kf][e]      = (_Float16)x0[e];
      aF[kf][e + 4]  = (_Float16)x1[e];
      aF[kf][e + 8]  = (_Float16)x2[e];
      aF[kf][e + 12] = (_Float16)x3[e];
    }
  }

  // ---- hidden = relu(X@gw1 + gb1) -> LDS (overwrite pos scratch) ----
#pragma unroll
  for (int nt = 0; nt < 8; nt++) {
    float bias = gb1L[16 * nt + lanelo];
    v8f acc;
#pragma unroll
    for (int i = 0; i < 8; i++) acc[i] = bias;
#pragma unroll
    for (int kf = 0; kf < 4; kf++) {
      v16h b = *(const v16h*)(smem + 1 * 32768 + ((nt * 4 + kf) * 32 + lane) * 32);
      acc = __builtin_amdgcn_wmma_f32_16x16x32_f16(false, aF[kf], false, b,
                                                   (short)0, acc, false, false);
    }
#pragma unroll
    for (int r = 0; r < 8; r++)
      posS[(r + grp * 8) * 128 + 16 * nt + lanelo] = fmaxf(acc[r], 0.f);
  }

  // ---- reload hidden as A fragments ----
#pragma unroll
  for (int kf = 0; kf < 4; kf++) {
    int c0 = 32 * kf + grp * 8;
    f4 h0 = *(const f4*)(prow + c0),      h1 = *(const f4*)(prow + c0 + 4);
    f4 h2 = *(const f4*)(prow + c0 + 16), h3 = *(const f4*)(prow + c0 + 20);
#pragma unroll
    for (int e = 0; e < 4; e++) {
      aF[kf][e]      = (_Float16)h0[e];
      aF[kf][e + 4]  = (_Float16)h1[e];
      aF[kf][e + 8]  = (_Float16)h2[e];
      aF[kf][e + 12] = (_Float16)h3[e];
    }
  }

  // ---- logits = h@gw2 + gb2; softmax over 17 slots; weighted sum ----
  const float* vbase = (const float*)(ws + WS_VFEAT) + (size_t)bb * NPTS * 128;
#pragma unroll
  for (int nt = 0; nt < 8; nt++) {
    int c = 16 * nt + lanelo;
    float bias = gb2L[c];
    v8f acc;
#pragma unroll
    for (int i = 0; i < 8; i++) acc[i] = bias;
#pragma unroll
    for (int kf = 0; kf < 4; kf++) {
      v16h b = *(const v16h*)(smem + 2 * 32768 + ((nt * 4 + kf) * 32 + lane) * 32);
      acc = __builtin_amdgcn_wmma_f32_16x16x32_f16(false, aF[kf], false, b,
                                                   (short)0, acc, false, false);
    }
    float gl = glL[c];
    float mx = acc[0];
#pragma unroll
    for (int r = 1; r < 8; r++) mx = fmaxf(mx, acc[r]);
    mx = fmaxf(mx, __shfl_xor(mx, 16, 32));   // other 8 rows live in lane^16
    mx = fmaxf(mx, gl);
    float e8[8], s = 0.f;
#pragma unroll
    for (int r = 0; r < 8; r++) { e8[r] = __expf(acc[r] - mx); s += e8[r]; }
    float eg = __expf(gl - mx);
    float stot = s + __shfl_xor(s, 16, 32) + eg;

    float red = 0.f;
#pragma unroll
    for (int r = 0; r < 8; r++) {
      int nidx = idxL[wave * 16 + r + grp * 8];
      float vv = vbase[(size_t)nidx * 128 + c];
      red += e8[r] * (vv + posr[nt][r]);
    }
    red += __shfl_xor(red, 16, 32);
    if (grp == 0) {
      red += eg * vgL[c];
      out[((size_t)bb * NQ + q) * 128 + c] = red / stot;
    }
  }
}

// =====================================================================
extern "C" void kernel_launch(void* const* d_in, const int* in_sizes, int n_in,
                              void* d_out, int out_size, void* d_ws, size_t ws_size,
                              hipStream_t stream) {
  const float* xyz_q  = (const float*)d_in[0];
  const float* lat    = (const float*)d_in[1];
  const float* xyz    = (const float*)d_in[2];
  const float* points = (const float*)d_in[3];
  const float* w_qs   = (const float*)d_in[4];
  const float* w_ks   = (const float*)d_in[5];
  const float* w_vs   = (const float*)d_in[6];
  const float* w_kg   = (const float*)d_in[7];
  const float* w_vg   = (const float*)d_in[8];
  const float* dw1    = (const float*)d_in[9];
  const float* db1    = (const float*)d_in[10];
  const float* dw2    = (const float*)d_in[11];
  const float* db2    = (const float*)d_in[12];
  const float* gw1    = (const float*)d_in[13];
  const float* gb1    = (const float*)d_in[14];
  const float* gw2    = (const float*)d_in[15];
  const float* gb2    = (const float*)d_in[16];
  char* ws = (char*)d_ws;
  float* out = (float*)d_out;

  pack_weights_kernel<<<dim3(5), 256, 0, stream>>>(w_ks, w_vs, dw2, gw1, gw2, ws);
  feat_gemm_kernel<<<dim3(64), 256, 0, stream>>>(points, ws);
  knn_kernel<<<dim3(16, 4), 256, 0, stream>>>(xyz_q, xyz, ws);
  globals_kernel<<<dim3(4), 128, 0, stream>>>(lat, w_qs, w_kg, w_vg,
                                              gw1, gb1, gw2, gb2, ws);
  cross_attn_kernel<<<dim3(512, 4), 256, LDS_TOTAL, stream>>>(
      xyz_q, xyz, dw1, db1, db2, gb1, gb2, ws, out);
}
